// HeadAttention_3229815406659
// MI455X (gfx1250) — compile-verified, matrix operand order checked by reading, not run on hardware
//
#include <hip/hip_runtime.h>

#define EMB   1024
#define HEAD  128
#define SEQ   4096
#define BATCH 4

typedef __attribute__((ext_vector_type(16))) __bf16 v16bf;
typedef __attribute__((ext_vector_type(8)))  __bf16 v8bf;
typedef __attribute__((ext_vector_type(8)))  float  v8f;

union Frag { v16bf v; v8bf h[2]; };

// Assemble a 16x32 bf16 A/B fragment for one lane: two contiguous 16B chunks.
// CDNA5 16-bit layout: lane group g=lane/16; halves 0..7 -> K = g*8 + h,
// halves 8..15 -> K = 16 + g*8 + (h-8).
static __device__ __forceinline__ v16bf load_frag(const __bf16* p_lo, const __bf16* p_hi) {
    Frag f;
    f.h[0] = *(const v8bf*)p_lo;
    f.h[1] = *(const v8bf*)p_hi;
    return f.v;
}

// CDNA5 async global->LDS copy, 16 bytes per lane, tracked by ASYNCcnt.
// Generic LDS pointers carry the LDS byte address in their low 32 bits.
static __device__ __forceinline__ void async_copy16(const __bf16* lds_ptr, const __bf16* gptr) {
    unsigned lo = (unsigned)(unsigned long long)lds_ptr;
    asm volatile("global_load_async_to_lds_b128 %0, %1, off"
                 :: "v"(lo), "v"(gptr) : "memory");
}
static __device__ __forceinline__ void wait_async0() {
    asm volatile("s_wait_asynccnt 0x0" ::: "memory");
}

// ---------------------------------------------------------------------------
// Prepass A: x (f32) -> xb (bf16), vectorized 8 elements/thread.
// ---------------------------------------------------------------------------
__global__ __launch_bounds__(256) void convert_x_kernel(const float* __restrict__ x,
                                                        __bf16* __restrict__ xb) {
    size_t i = ((size_t)blockIdx.x * 256 + threadIdx.x) * 8;
    float4 a = *(const float4*)(x + i);
    float4 b = *(const float4*)(x + i + 4);
    v8bf o;
    o[0] = (__bf16)a.x; o[1] = (__bf16)a.y; o[2] = (__bf16)a.z; o[3] = (__bf16)a.w;
    o[4] = (__bf16)b.x; o[5] = (__bf16)b.y; o[6] = (__bf16)b.z; o[7] = (__bf16)b.w;
    *(v8bf*)(xb + i) = o;
}

// ---------------------------------------------------------------------------
// Prepass B: W[E][H] f32 -> WT[H][E] bf16 (B-operand layout), per matrix.
// ---------------------------------------------------------------------------
__global__ __launch_bounds__(256) void transpose_w_kernel(
    const float* __restrict__ Wq, const float* __restrict__ Wk, const float* __restrict__ Wv,
    __bf16* __restrict__ wt_all) {
    const int which = blockIdx.y;
    const float* W = (which == 0) ? Wq : (which == 1) ? Wk : Wv;
    __bf16* WT = wt_all + (size_t)which * HEAD * EMB;
    int o = blockIdx.x * 256 + threadIdx.x;      // over H*E, coalesced writes
    int h = o / EMB, k = o % EMB;
    WT[o] = (__bf16)W[(size_t)k * HEAD + h];
}

// ---------------------------------------------------------------------------
// Kernel 1: q/k/v = x @ W + b (bf16 WMMA, f32 accum), async double-buffered
// LDS staging. Block = 128 rows x 128 cols; blockIdx.y selects q/k/v.
// ---------------------------------------------------------------------------
__global__ __launch_bounds__(256) void qkv_proj_kernel(
    const __bf16* __restrict__ xb, const __bf16* __restrict__ wt_all,
    const float* __restrict__ bq, const float* __restrict__ bk, const float* __restrict__ bv,
    __bf16* __restrict__ qo, __bf16* __restrict__ ko, __bf16* __restrict__ vo)
{
    constexpr int XS = 40;                       // padded LDS stride (halves)
    __shared__ __bf16 xt[2][128 * XS];           // x tile  [row][k]
    __shared__ __bf16 wt[2][128 * XS];           // W tile  [n][k]

    const int which = blockIdx.y;
    const float* bias = (which == 0) ? bq : (which == 1) ? bk : bv;
    __bf16*      out  = (which == 0) ? qo : (which == 1) ? ko : vo;
    const __bf16* WT  = wt_all + (size_t)which * HEAD * EMB;

    const int t    = threadIdx.x;
    const int lane = t & 31;
    const int w    = t >> 5;
    const int l16  = lane & 15;
    const int g    = lane >> 4;
    const int row0 = blockIdx.x * 128;

    // stage one 128x32 bf16 tile of x and of W^T via async b128 copies
    auto stage = [&](int buf, int kb) {
        for (int i = 0; i < 2; ++i) {
            int c = t + i * 256;                 // 512 chunks of 8 halves
            int r = c >> 2, seg = (c & 3) * 8;
            async_copy16(&xt[buf][r * XS + seg], xb + (size_t)(row0 + r) * EMB + kb + seg);
            async_copy16(&wt[buf][r * XS + seg], WT + (size_t)r * EMB + kb + seg);
        }
    };

    v8f acc[8] = {};
    stage(0, 0);
    for (int s = 0; s < EMB / 32; ++s) {
        const int buf = s & 1;
        wait_async0();
        __syncthreads();
        if (s + 1 < EMB / 32) stage(buf ^ 1, (s + 1) * 32);

        const __bf16* xr = &xt[buf][(w * 16 + l16) * XS];
        v16bf a = load_frag(xr + g * 8, xr + 16 + g * 8);
        for (int n = 0; n < 8; ++n) {
            const __bf16* wr = &wt[buf][(n * 16 + l16) * XS];
            v16bf bf = load_frag(wr + g * 8, wr + 16 + g * 8);
            acc[n] = __builtin_amdgcn_wmma_f32_16x16x32_bf16(
                false, a, false, bf, (short)0, acc[n], false, false);
        }
    }

    for (int n = 0; n < 8; ++n) {
        float bn = bias[n * 16 + l16];
        for (int r = 0; r < 8; ++r) {
            int row = row0 + w * 16 + r + 8 * g;
            out[(size_t)row * HEAD + n * 16 + l16] = (__bf16)(acc[n][r] + bn);
        }
    }
}

// ---------------------------------------------------------------------------
// Kernel 2: causal flash attention, async double-buffered K tiles,
// exp2-domain online softmax, bf16 WMMA with f32 accumulation.
// ---------------------------------------------------------------------------
__global__ __launch_bounds__(256) void flash_attn_kernel(
    const __bf16* __restrict__ qw, const __bf16* __restrict__ kw,
    const __bf16* __restrict__ vw, float* __restrict__ out)
{
    constexpr int KS = 136;                      // K tile stride (halves)
    constexpr int VS = 40;                       // V/P tile stride (halves)
    __shared__ __bf16 kt[2][32 * KS];            // K tile  [kv][h]  (B for QK^T)
    __shared__ __bf16 vt[2][HEAD * VS];          // V tile  [h][kv]  (B for PV)
    __shared__ __bf16 pt[8 * 16 * VS];           // per-wave P scratch [m][kv]

    const int t    = threadIdx.x;
    const int lane = t & 31;
    const int w    = t >> 5;
    const int l16  = lane & 15;
    const int g    = lane >> 4;
    const int b    = blockIdx.y;
    const int q_base = blockIdx.x * 128;
    const int qb   = q_base + w * 16;

    // Q fragments: 16x128 bf16 = 4 A-frags, kept in registers
    const __bf16* qrow = qw + ((size_t)b * SEQ + qb + l16) * HEAD;
    v16bf qa[4];
    for (int c = 0; c < 4; ++c)
        qa[c] = load_frag(qrow + c * 32 + g * 8, qrow + c * 32 + 16 + g * 8);

    v8f   O[8] = {};
    float m[8], l[8];
    for (int r = 0; r < 8; ++r) { m[r] = -__builtin_inff(); l[r] = 0.0f; }

    // scores pre-scaled into log2 domain: (1/sqrt(128)) * log2(e)
    const float kscale = 0.1275203082f;
    const int nsteps = (q_base >> 5) + 4;        // causal bound

    // stage one 32x128 K tile (async) + transposed V tile (vectorized scatter)
    auto stage_kv = [&](int buf, int kvb) {
        for (int i = 0; i < 2; ++i) {
            int c = t + i * 256;                 // 512 chunks of 8 halves
            int kv = c >> 4, seg = (c & 15) * 8;
            size_t src = ((size_t)b * SEQ + kvb + kv) * HEAD + seg;
            async_copy16(&kt[buf][kv * KS + seg], kw + src);
            v8bf vv = *(const v8bf*)(vw + src);
            for (int j = 0; j < 8; ++j)
                vt[buf][(seg + j) * VS + kv] = vv[j];
        }
    };

    stage_kv(0, 0);
    for (int s = 0; s < nsteps; ++s) {
        const int buf = s & 1;
        const int kvb = s * 32;
        wait_async0();
        __syncthreads();
        if (s + 1 < nsteps) stage_kv(buf ^ 1, (s + 1) * 32);

        // scores: two 16x16 accums, K = H = 128 (4 WMMA each)
        v8f sc[2] = {};
        for (int n = 0; n < 2; ++n) {
            const __bf16* kr = &kt[buf][(n * 16 + l16) * KS];
            for (int c = 0; c < 4; ++c) {
                v16bf bf = load_frag(kr + c * 32 + g * 8, kr + c * 32 + 16 + g * 8);
                sc[n] = __builtin_amdgcn_wmma_f32_16x16x32_bf16(
                    false, qa[c], false, bf, (short)0, sc[n], false, false);
            }
        }

        // causal mask + online softmax (exp2 domain)
        __bf16* pw = &pt[(w * 16) * VS];
        for (int r = 0; r < 8; ++r) {
            int mrow = r + 8 * g;
            int grow = qb + mrow;
            float s0 = sc[0][r] * kscale;
            float s1 = sc[1][r] * kscale;
            if (kvb + l16 > grow)      s0 = -__builtin_inff();
            if (kvb + 16 + l16 > grow) s1 = -__builtin_inff();
            float mt = fmaxf(s0, s1);
            for (int off = 8; off > 0; off >>= 1)
                mt = fmaxf(mt, __shfl_xor(mt, off, 16));
            float mn    = fmaxf(m[r], mt);
            float alpha = exp2f(m[r] - mn);
            float p0 = exp2f(s0 - mn);
            float p1 = exp2f(s1 - mn);
            float ps = p0 + p1;
            for (int off = 8; off > 0; off >>= 1)
                ps += __shfl_xor(ps, off, 16);
            l[r] = l[r] * alpha + ps;
            m[r] = mn;
            for (int n = 0; n < 8; ++n) O[n][r] = O[n][r] * alpha;
            pw[mrow * VS + l16]      = (__bf16)p0;
            pw[mrow * VS + 16 + l16] = (__bf16)p1;
        }

        // O += P (16x32) x V (32x128)
        const __bf16* pr = &pt[(w * 16 + l16) * VS];
        v16bf pa = load_frag(pr + g * 8, pr + 16 + g * 8);
        for (int n = 0; n < 8; ++n) {
            const __bf16* vr = &vt[buf][(n * 16 + l16) * VS];
            v16bf vb = load_frag(vr + g * 8, vr + 16 + g * 8);
            O[n] = __builtin_amdgcn_wmma_f32_16x16x32_bf16(
                false, pa, false, vb, (short)0, O[n], false, false);
        }
        __syncthreads();
    }

    for (int r = 0; r < 8; ++r) {
        int grow = qb + r + 8 * g;
        float inv = 1.0f / l[r];
        for (int n = 0; n < 8; ++n)
            out[((size_t)b * SEQ + grow) * HEAD + n * 16 + l16] = O[n][r] * inv;
    }
}

// ---------------------------------------------------------------------------
extern "C" void kernel_launch(void* const* d_in, const int* in_sizes, int n_in,
                              void* d_out, int out_size, void* d_ws, size_t ws_size,
                              hipStream_t stream) {
    const float* x  = (const float*)d_in[0];
    const float* Wq = (const float*)d_in[1];
    const float* Wk = (const float*)d_in[2];
    const float* Wv = (const float*)d_in[3];
    const float* bq = (const float*)d_in[4];
    const float* bk = (const float*)d_in[5];
    const float* bv = (const float*)d_in[6];
    float* out = (float*)d_out;

    const size_t qkv_elems = (size_t)BATCH * SEQ * HEAD;      // 2 MiB elems
    const size_t x_elems   = (size_t)BATCH * SEQ * EMB;       // 16 Mi elems
    __bf16* qo = (__bf16*)d_ws;
    __bf16* ko = qo + qkv_elems;
    __bf16* vo = ko + qkv_elems;
    __bf16* xb = vo + qkv_elems;
    __bf16* wtT = xb + x_elems;                               // 3 x [H][E]

    convert_x_kernel<<<dim3(x_elems / (256 * 8)), 256, 0, stream>>>(x, xb);
    transpose_w_kernel<<<dim3(EMB * HEAD / 256, 3), 256, 0, stream>>>(Wq, Wk, Wv, wtT);

    dim3 gproj(BATCH * SEQ / 128, 3);
    qkv_proj_kernel<<<gproj, 256, 0, stream>>>(xb, wtT, bq, bk, bv, qo, ko, vo);

    dim3 gattn(SEQ / 128, BATCH);
    flash_attn_kernel<<<gattn, 256, 0, stream>>>(qo, ko, vo, out);
}
